// DeformableMask_26199300505715
// MI455X (gfx1250) — matrix operand order repaired and checked by reading, not compiled, not run on hardware
//
#include <hip/hip_runtime.h>
#include <hip/hip_bf16.h>
#include <math.h>

// ---------------------------------------------------------------------------
// Shapes (compile-time constants from the reference)
// ---------------------------------------------------------------------------
#define BATCH   4
#define CIN     256
#define COUT    256
#define HH      96
#define WW      96
#define HW      (HH * WW)          // 9216
#define NPOS    (BATCH * HW)       // 36864
#define K2      9
#define KTOT    (K2 * CIN)         // 2304
#define NKC     (KTOT / 32)        // 72 K-chunks of 32 (tap-major ordering)
#define MTILE   32
#define NTILES_M (NPOS / MTILE)    // 1152

typedef __attribute__((ext_vector_type(16))) __bf16 v16bf;
typedef __attribute__((ext_vector_type(8)))  float  v8f;

__device__ __forceinline__ unsigned short f2bfu(float f) {
    unsigned int u = __float_as_uint(f);
    unsigned int r = (u + 0x7FFFu + ((u >> 16) & 1u)) >> 16;   // round-nearest-even
    return (unsigned short)r;
}

// ---------------------------------------------------------------------------
// K0: pack w_dcn (Cout, Cin, 3, 3) into per-lane WMMA B fragments (bf16 pairs)
// Global K ordering: k = tap*256 + cin  (so each 32-chunk is a single tap).
// B fragment layout (32x16, 16-bit): lane n = l&15;
//   lanes 0-15 : VGPR v holds K = 2v, 2v+1
//   lanes 16-31: VGPR v holds K = 16+2v, 16+2v+1
// Storage: bpack[((kc*16 + nb)*32 + lane)*8 + v]
// ---------------------------------------------------------------------------
__global__ __launch_bounds__(256) void pack_b(const float* __restrict__ wd,
                                              unsigned int* __restrict__ bpack) {
    int idx = blockIdx.x * 256 + threadIdx.x;          // 72*16*32*8 = 294912
    if (idx >= NKC * 16 * 32 * 8) return;
    int v    = idx & 7;
    int lane = (idx >> 3) & 31;
    int tile = idx >> 8;                               // kc*16 + nb
    int nb   = tile & 15;
    int kc   = tile >> 4;
    int n    = nb * 16 + (lane & 15);
    int kbase = 2 * v + ((lane >> 4) ? 16 : 0);
    int kg   = kc * 32 + kbase;                        // tap-major global k
    int tap  = kg >> 8;
    int cin  = kg & 255;                               // even
    float lo = wd[n * KTOT + cin * 9 + tap];
    float hi = wd[n * KTOT + (cin + 1) * 9 + tap];
    bpack[idx] = (unsigned)f2bfu(lo) | ((unsigned)f2bfu(hi) << 16);
}

// ---------------------------------------------------------------------------
// K1: offset conv — x:(4,256,96,96) * w_off:(18,256,3,3) -> offs:(4,18,96,96)
// One block per (b,y) row; 96 threads over x; 18 accumulators per thread.
// ---------------------------------------------------------------------------
__global__ __launch_bounds__(96) void offset_conv(const float* __restrict__ x,
                                                  const float* __restrict__ woff,
                                                  const float* __restrict__ boff,
                                                  float* __restrict__ offs) {
    int row = blockIdx.x;                 // b*96 + y
    int b = row / HH, y = row % HH;
    int xx = threadIdx.x;
    float acc[18];
#pragma unroll
    for (int o = 0; o < 18; ++o) acc[o] = boff[o];
    for (int ci = 0; ci < CIN; ++ci) {
        const float* img = x + (b * CIN + ci) * HW;
#pragma unroll
        for (int ky = 0; ky < 3; ++ky) {
            int yy = y + ky - 1;
            if (yy < 0 || yy >= HH) continue;
#pragma unroll
            for (int kx = 0; kx < 3; ++kx) {
                int xs = xx + kx - 1;
                int xc = min(max(xs, 0), WW - 1);
                float msk = (xs >= 0 && xs < WW) ? 1.f : 0.f;
                float v = img[yy * WW + xc] * msk;
                const float* wp = woff + ci * 9 + ky * 3 + kx;
#pragma unroll
                for (int o = 0; o < 18; ++o)
                    acc[o] = fmaf(v, wp[o * KTOT], acc[o]);
            }
        }
    }
#pragma unroll
    for (int o = 0; o < 18; ++o)
        offs[(b * 18 + o) * HW + y * WW + xx] = acc[o];
}

// ---------------------------------------------------------------------------
// K2: deformable-conv GEMM via WMMA bf16.
// Block = 256 threads (8 waves). Tile: 32 positions x 256 Cout.
// Wave w: mb = w>>2 (M sub-tile), owns 4 N-tiles (w&3)*4 .. +3.
// K loop: 72 chunks of 32 (tap = kc/8, cin0 = (kc&7)*32).
// ---------------------------------------------------------------------------
__global__ __launch_bounds__(256) void deform_gemm(const float* __restrict__ x,
                                                   const float* __restrict__ offs,
                                                   const unsigned int* __restrict__ bpack,
                                                   const float* __restrict__ bias,
                                                   float* __restrict__ x1) {
    __shared__ unsigned short Alds[MTILE][34];   // 32x32 bf16, padded row (gcd(17,32)=1)
    __shared__ int   siy0[MTILE], six0[MTILE], siy1[MTILE], six1[MTILE];
    __shared__ float sw00[MTILE], sw01[MTILE], sw10[MTILE], sw11[MTILE];

    const int tid  = threadIdx.x;
    const int lane = tid & 31;
    const int wave = tid >> 5;
    const int mb   = wave >> 2;           // 0..1
    const int nb0  = (wave & 3) * 4;      // 0,4,8,12

    const int tile = blockIdx.x;          // 0..1151
    const int p0   = tile * MTILE;
    const int b    = p0 / HW;             // 9216 % 32 == 0 -> tile never crosses batch
    const int sp0  = p0 % HW;

    v8f acc[4];
#pragma unroll
    for (int j = 0; j < 4; ++j) acc[j] = (v8f){0.f,0.f,0.f,0.f,0.f,0.f,0.f,0.f};

    for (int kc = 0; kc < NKC; ++kc) {
        const int tap  = kc >> 3;
        const int cin0 = (kc & 7) << 5;

        // --- bilinear coords: recompute when tap changes (every 8 chunks) ---
        if ((kc & 7) == 0 && tid < MTILE) {
            int m  = tid;
            int sp = sp0 + m;
            int yy = sp / WW, xq = sp % WW;
            int ky = tap / 3 - 1, kx = tap % 3 - 1;
            const float* ob = offs + (b * 18 + tap * 2) * HW;
            float py = (float)(yy + ky) + ob[sp];
            float px = (float)(xq + kx) + ob[HW + sp];
            float fy = floorf(py), fx = floorf(px);
            float wy = py - fy,    wx = px - fx;
            int y0 = (int)fy, x0 = (int)fx;
            int y1 = y0 + 1,  xo = x0 + 1;
            float vy0 = (y0 >= 0 && y0 < HH) ? 1.f : 0.f;
            float vy1 = (y1 >= 0 && y1 < HH) ? 1.f : 0.f;
            float vx0 = (x0 >= 0 && x0 < WW) ? 1.f : 0.f;
            float vx1 = (xo >= 0 && xo < WW) ? 1.f : 0.f;
            sw00[m] = (1.f - wy) * (1.f - wx) * vy0 * vx0;
            sw01[m] = (1.f - wy) * wx         * vy0 * vx1;
            sw10[m] = wy         * (1.f - wx) * vy1 * vx0;
            sw11[m] = wy         * wx         * vy1 * vx1;
            siy0[m] = min(max(y0, 0), HH - 1);
            siy1[m] = min(max(y1, 0), HH - 1);
            six0[m] = min(max(x0, 0), WW - 1);
            six1[m] = min(max(xo, 0), WW - 1);
        }
        __syncthreads();   // coords visible; prior WMMA reads of Alds complete

        // --- gather + lerp 32x32 A tile into LDS as bf16 ---
        {
            int e = tid;
#pragma unroll
            for (int i = 0; i < 4; ++i, e += 256) {
                int m = e & 31;        // consecutive tids -> consecutive positions
                int c = e >> 5;
                const float* img = x + (b * CIN + cin0 + c) * HW;
                int iy0 = siy0[m], ix0 = six0[m], iy1 = siy1[m], ix1 = six1[m];
                float v = sw00[m] * img[iy0 * WW + ix0]
                        + sw01[m] * img[iy0 * WW + ix1]
                        + sw10[m] * img[iy1 * WW + ix0]
                        + sw11[m] * img[iy1 * WW + ix1];
                Alds[m][c] = f2bfu(v);
            }
        }
        __syncthreads();   // A tile ready

        // --- build A fragment (16-bit A layout, 16x32) ---
        const int mrow = mb * 16 + (lane & 15);
        const int koff = (lane >> 4) ? 16 : 0;
        union { v16bf v; unsigned int u[8]; } afrag;
#pragma unroll
        for (int vv = 0; vv < 8; ++vv) {
            int k = 2 * vv + koff;   // even
            afrag.u[vv] = *(const unsigned int*)&Alds[mrow][k];
        }

        // --- 4 WMMAs against prepacked B fragments ---
#pragma unroll
        for (int j = 0; j < 4; ++j) {
            int nb = nb0 + j;
            const uint4* bp = (const uint4*)(bpack + (((kc * 16 + nb) * 32 + lane) << 3));
            union { v16bf v; uint4 q[2]; } bfrag;
            bfrag.q[0] = bp[0];
            bfrag.q[1] = bp[1];
            acc[j] = __builtin_amdgcn_wmma_f32_16x16x32_bf16(
                false, afrag.v, false, bfrag.v, (short)0, acc[j], false, false);
        }
        __syncthreads();   // done reading Alds before next fill
    }

    // --- epilogue: D layout -> x1_raw (+bias). m = mb*16 + r + (lane>=16?8:0)
#pragma unroll
    for (int j = 0; j < 4; ++j) {
        int n  = (nb0 + j) * 16 + (lane & 15);
        float bv = bias[n];
        int mofs = mb * 16 + ((lane >> 4) ? 8 : 0);
        float* outp = x1 + (b * COUT + n) * HW + sp0 + mofs;
#pragma unroll
        for (int r = 0; r < 8; ++r)
            outp[r] = acc[j][r] + bv;
    }
}

// ---------------------------------------------------------------------------
// K3: GroupNorm stats — block per (b, g); reduce 8*9216 elems
// ---------------------------------------------------------------------------
__global__ __launch_bounds__(256) void gn_stats(const float* __restrict__ x1,
                                                float* __restrict__ stats) {
    int bg = blockIdx.x;                  // b*32 + g
    int b = bg >> 5, g = bg & 31;
    const float* base = x1 + (b * COUT + g * 8) * HW;
    float s = 0.f, ss = 0.f;
    for (int i = threadIdx.x; i < 8 * HW; i += 256) {
        float v = base[i];
        s += v; ss += v * v;
    }
    __shared__ float rs[256], rss[256];
    rs[threadIdx.x] = s; rss[threadIdx.x] = ss;
    __syncthreads();
    for (int o = 128; o > 0; o >>= 1) {
        if (threadIdx.x < o) {
            rs[threadIdx.x]  += rs[threadIdx.x + o];
            rss[threadIdx.x] += rss[threadIdx.x + o];
        }
        __syncthreads();
    }
    if (threadIdx.x == 0) {
        float n   = (float)(8 * HW);
        float mu  = rs[0] / n;
        float var = rss[0] / n - mu * mu;
        stats[bg * 2]     = mu;
        stats[bg * 2 + 1] = rsqrtf(var + 1e-5f);
    }
}

// ---------------------------------------------------------------------------
// K4: GroupNorm apply (in place)
// ---------------------------------------------------------------------------
__global__ __launch_bounds__(256) void gn_apply(float* __restrict__ x1,
                                                const float* __restrict__ stats,
                                                const float* __restrict__ gamma,
                                                const float* __restrict__ beta) {
    int i = blockIdx.x * 256 + threadIdx.x;
    if (i >= BATCH * COUT * HW) return;
    int bc = i / HW;
    int c  = bc & 255;
    int b  = bc >> 8;
    int g  = c >> 3;
    float mu   = stats[(b * 32 + g) * 2];
    float rstd = stats[(b * 32 + g) * 2 + 1];
    x1[i] = (x1[i] - mu) * rstd * gamma[c] + beta[c];
}

// ---------------------------------------------------------------------------
// K5: mask conv (1 out channel) + sigmoid -> s:(4,96,96)
// ---------------------------------------------------------------------------
__global__ __launch_bounds__(96) void mask_conv(const float* __restrict__ x1,
                                                const float* __restrict__ w2,
                                                const float* __restrict__ b2,
                                                float* __restrict__ s) {
    int row = blockIdx.x;                 // b*96 + y
    int b = row / HH, y = row % HH;
    int xx = threadIdx.x;
    float acc = b2[0];
    for (int ci = 0; ci < COUT; ++ci) {
        const float* img = x1 + (b * COUT + ci) * HW;
        const float* wp  = w2 + ci * 9;
#pragma unroll
        for (int ky = 0; ky < 3; ++ky) {
            int yy = y + ky - 1;
            if (yy < 0 || yy >= HH) continue;
#pragma unroll
            for (int kx = 0; kx < 3; ++kx) {
                int xs = xx + kx - 1;
                int xc = min(max(xs, 0), WW - 1);
                float msk = (xs >= 0 && xs < WW) ? 1.f : 0.f;
                acc = fmaf(img[yy * WW + xc] * msk, wp[ky * 3 + kx], acc);
            }
        }
    }
    s[row * WW + xx] = 1.f / (1.f + __expf(-acc));
}

// ---------------------------------------------------------------------------
// K6: out = x1 * s (broadcast over channels)
// ---------------------------------------------------------------------------
__global__ __launch_bounds__(256) void final_mul(const float* __restrict__ x1,
                                                 const float* __restrict__ s,
                                                 float* __restrict__ out) {
    int i = blockIdx.x * 256 + threadIdx.x;
    if (i >= BATCH * COUT * HW) return;
    int sp = i % HW;
    int b  = i / (COUT * HW);
    out[i] = x1[i] * s[b * HW + sp];
}

// ---------------------------------------------------------------------------
// Launcher
// ---------------------------------------------------------------------------
extern "C" void kernel_launch(void* const* d_in, const int* in_sizes, int n_in,
                              void* d_out, int out_size, void* d_ws, size_t ws_size,
                              hipStream_t stream) {
    const float* x     = (const float*)d_in[0];
    const float* w_off = (const float*)d_in[1];
    const float* b_off = (const float*)d_in[2];
    const float* w_dcn = (const float*)d_in[3];
    const float* b_dcn = (const float*)d_in[4];
    const float* gamma = (const float*)d_in[5];
    const float* beta  = (const float*)d_in[6];
    const float* w2    = (const float*)d_in[7];
    const float* b2    = (const float*)d_in[8];
    float* out = (float*)d_out;
    float* ws  = (float*)d_ws;

    // workspace layout (float offsets)
    float*        offs  = ws;                                   // 663552
    unsigned int* bpack = (unsigned int*)(ws + 663552);         // 294912 u32
    float*        stats = ws + 663552 + 294912;                 // 256
    float*        x1    = ws + 958720;                          // 9437184
    float*        sbuf  = ws + 10395904;                        // 36864

    const int NELEM = BATCH * COUT * HW;                        // 9437184

    pack_b     <<<(NKC * 16 * 32 * 8 + 255) / 256, 256, 0, stream>>>(w_dcn, bpack);
    offset_conv<<<BATCH * HH, 96, 0, stream>>>(x, w_off, b_off, offs);
    deform_gemm<<<NTILES_M, 256, 0, stream>>>(x, offs, bpack, b_dcn, x1);
    gn_stats   <<<BATCH * 32, 256, 0, stream>>>(x1, stats);
    gn_apply   <<<(NELEM + 255) / 256, 256, 0, stream>>>(x1, stats, gamma, beta);
    mask_conv  <<<BATCH * HH, 96, 0, stream>>>(x1, w2, b2, sbuf);
    final_mul  <<<(NELEM + 255) / 256, 256, 0, stream>>>(x1, sbuf, out);
}
